// JanusSelfAttention_45526653337850
// MI455X (gfx1250) — matrix operand order repaired and checked by reading, not compiled, not run on hardware
//
#include <hip/hip_runtime.h>

// ---------------------------------------------------------------------------
// MI455X (gfx1250, wave32) implementation of Janus self-attention.
// f16 WMMA (v_wmma_f32_16x16x32_f16) with fp32 accumulation everywhere.
// ---------------------------------------------------------------------------

typedef _Float16 h4   __attribute__((ext_vector_type(4)));
typedef _Float16 h8   __attribute__((ext_vector_type(8)));
typedef _Float16 v16h __attribute__((ext_vector_type(16)));
typedef float    v8f  __attribute__((ext_vector_type(8)));
typedef float    f4   __attribute__((ext_vector_type(4)));

#define B_   2
#define S_   2048
#define D_   1024
#define H_   16
#define KH_  4
#define HD_  64
#define WIN_ 512

__device__ __forceinline__ v8f wmma16(v16h a, v16h b, v8f c) {
  // D = A(16x32 f16) * B(32x16 f16) + C(16x16 f32)
  return __builtin_amdgcn_wmma_f32_16x16x32_f16(false, a, false, b,
                                                (short)0, c, false, false);
}

// Load a 16-half fragment as two 16B vector loads (alignment-safe).
__device__ __forceinline__ v16h ldfrag16(const _Float16* p0, const _Float16* p1) {
  h8 lo = *(const h8*)p0;
  h8 hi = *(const h8*)p1;
  return __builtin_shufflevector(lo, hi, 0,1,2,3,4,5,6,7,8,9,10,11,12,13,14,15);
}

__device__ __forceinline__ h4 cvt4(f4 v) {
  h4 r;
  r[0] = (_Float16)v[0]; r[1] = (_Float16)v[1];
  r[2] = (_Float16)v[2]; r[3] = (_Float16)v[3];
  return r;
}

// ---------------------------------------------------------------------------
// Generic GEMM: C[M,N] = A[M,K] * W[N,K]^T   (fp32 global, f16 WMMA compute)
// Block tile 128x128, 8 waves, each wave: 16 rows x 8 N-tiles of 16.
// Double-buffered LDS; vectorized (b128) global staging; packed b64 LDS stores.
// ---------------------------------------------------------------------------
#define BM 128
#define BN 128
#define BK 32

__global__ __launch_bounds__(256) void k_gemm(const float* __restrict__ A,
                                              const float* __restrict__ W,
                                              float* __restrict__ C,
                                              int M, int N, int K) {
  __shared__ __attribute__((aligned(32))) _Float16 As[2 * BM * BK];
  __shared__ __attribute__((aligned(32))) _Float16 Ws[2 * BN * BK];

  const int m0   = blockIdx.x * BM;
  const int n0   = blockIdx.y * BN;
  const int tid  = threadIdx.x;
  const int wave = tid >> 5;
  const int lane = tid & 31;

  const int mrow = wave * 16 + (lane & 15);   // A-fragment row
  const int hi   = lane >> 4;
  const int koff = hi * 8;                    // A-fragment K split
  const int kb   = hi * 8;                    // B-fragment K split
  const int nn   = lane & 15;                 // B-fragment column

  // Per-thread staging: 4 float4 from A tile (128x32), 4 float4 from W tile
  // (128x32). float4 index f -> row = f>>3, col = (f&7)*4; LDS dest = f*4.
  f4 areg[4], wreg[4];

  auto stage_load = [&](int kc) {
    #pragma unroll
    for (int i = 0; i < 4; ++i) {
      int f = tid + i * 256;
      int r = f >> 3, c = (f & 7) * 4;
      areg[i] = *(const f4*)(A + (m0 + r) * K + kc + c);
    }
    #pragma unroll
    for (int i = 0; i < 4; ++i) {
      int f = tid + i * 256;
      int r = f >> 3, c = (f & 7) * 4;
      wreg[i] = *(const f4*)(W + (n0 + r) * K + kc + c);
    }
  };
  auto stage_store = [&](int buf) {
    _Float16* as = As + buf * (BM * BK);
    _Float16* ws = Ws + buf * (BN * BK);
    #pragma unroll
    for (int i = 0; i < 4; ++i) {
      int f = tid + i * 256;
      *(h4*)(as + f * 4) = cvt4(areg[i]);
    }
    #pragma unroll
    for (int i = 0; i < 4; ++i) {
      int f = tid + i * 256;
      *(h4*)(ws + f * 4) = cvt4(wreg[i]);
    }
  };

  v8f acc[8];
  #pragma unroll
  for (int j = 0; j < 8; ++j) acc[j] = (v8f){};

  stage_load(0);
  stage_store(0);
  int buf = 0;

  for (int kc = 0; kc < K; kc += BK) {
    const bool more = (kc + BK) < K;
    if (more) stage_load(kc + BK);          // issue next chunk's global loads
    if (kc + 2 * BK < K) {                  // prefetch chunk after that
      __builtin_prefetch(A + (m0 + (tid >> 1)) * K + kc + 2 * BK + (tid & 1) * 16, 0, 1);
      __builtin_prefetch(W + (n0 + (tid >> 1)) * K + kc + 2 * BK + (tid & 1) * 16, 0, 1);
    }
    __syncthreads();                        // LDS[buf] ready for everyone

    const _Float16* as = As + buf * (BM * BK);
    const _Float16* ws = Ws + buf * (BN * BK);
    // A fragment (16x32, ISA A-layout) from LDS.
    v16h a = ldfrag16(as + mrow * 32 + koff, as + mrow * 32 + 16 + koff);
    // B fragments (32x16, ISA B-layout): lane's column is contiguous in Ws.
    #pragma unroll
    for (int j = 0; j < 8; ++j) {
      v16h b = ldfrag16(ws + (j * 16 + nn) * 32 + kb,
                        ws + (j * 16 + nn) * 32 + kb + 16);
      acc[j] = wmma16(a, b, acc[j]);
    }

    if (more) stage_store(buf ^ 1);         // waits loads; fills other buffer
    buf ^= 1;
  }

  // C layout: v8f elem r, lane l -> (M = r + 8*(l>>4), N = l&15)
  #pragma unroll
  for (int r = 0; r < 8; ++r) {
    int row = (m0 + wave * 16 + r + 8 * hi) * N + n0;
    #pragma unroll
    for (int j = 0; j < 8; ++j) C[row + j * 16 + nn] = acc[j][r];
  }
}

// ---------------------------------------------------------------------------
// Fused RMSNorm + RoPE for q and k; transposes to [B,H(KH),S,HD] f16.
// One wave32 per 64-element head row; lane j owns RoPE pair (2j, 2j+1).
// q additionally folds in softmax_scale * log2(e).
// ---------------------------------------------------------------------------
__global__ __launch_bounds__(256) void k_normrope(const float* __restrict__ xq,
                                                  const float* __restrict__ xk,
                                                  const float* __restrict__ fc,
                                                  const float* __restrict__ fs,
                                                  const float* __restrict__ qw,
                                                  const float* __restrict__ kw,
                                                  _Float16* __restrict__ q16,
                                                  _Float16* __restrict__ k16) {
  const int wid  = blockIdx.x * 8 + (threadIdx.x >> 5);
  const int lane = threadIdx.x & 31;
  const int QROWS = B_ * S_ * H_;
  const float QSCALE = 0.125f * 1.44269504088896f;  // 1/sqrt(64) * log2(e)

  const float* src; const float* w; _Float16* dst;
  float oscale; int spos;
  if (wid < QROWS) {
    int bs = wid / H_, h = wid - bs * H_;
    int b = bs / S_, s = bs - b * S_;
    src = xq + bs * (H_ * HD_) + h * HD_;
    dst = q16 + ((b * H_ + h) * S_ + s) * HD_;
    w = qw; oscale = QSCALE; spos = s;
  } else {
    int wid2 = wid - QROWS;
    int bs = wid2 / KH_, kh = wid2 - bs * KH_;
    int b = bs / S_, s = bs - b * S_;
    src = xk + bs * (KH_ * HD_) + kh * HD_;
    dst = k16 + ((b * KH_ + kh) * S_ + s) * HD_;
    w = kw; oscale = 1.0f; spos = s;
  }

  float xr = src[2 * lane], xi = src[2 * lane + 1];
  float ss = xr * xr + xi * xi;
  #pragma unroll
  for (int m = 1; m < 32; m <<= 1) ss += __shfl_xor(ss, m, 32);
  float scale = rsqrtf(ss * (1.0f / HD_) + 1e-5f);
  float nr = xr * scale * w[2 * lane];
  float ni = xi * scale * w[2 * lane + 1];
  float c  = fc[spos * (HD_ / 2) + lane];
  float sn = fs[spos * (HD_ / 2) + lane];
  dst[2 * lane]     = (_Float16)((nr * c - ni * sn) * oscale);
  dst[2 * lane + 1] = (_Float16)((nr * sn + ni * c) * oscale);
}

// ---------------------------------------------------------------------------
// V transpose+convert: xv [B*S, KH*HD] fp32 -> vT [B,KH,HD,S] f16
// so attention B-fragments for P*V are contiguous 32B loads.
// ---------------------------------------------------------------------------
__global__ __launch_bounds__(256) void k_vtrans(const float* __restrict__ xv,
                                                _Float16* __restrict__ vT) {
  int g  = blockIdx.x * 256 + threadIdx.x;
  int s  = g & (S_ - 1);
  int d  = (g >> 11) & (HD_ - 1);
  int kh = (g >> 17) & (KH_ - 1);
  int b  = g >> 19;
  vT[g] = (_Float16)xv[(b * S_ + s) * (KH_ * HD_) + kh * HD_ + d];
}

// ---------------------------------------------------------------------------
// Sliding-window flash attention. One wave per 16-query tile; 32-key blocks.
// q in log2-exp domain (scale folded), online softmax on C-layout accums.
// ---------------------------------------------------------------------------
__global__ __launch_bounds__(128) void k_attn(const _Float16* __restrict__ q16,
                                              const _Float16* __restrict__ k16,
                                              const _Float16* __restrict__ vT,
                                              float* __restrict__ ao) {
  __shared__ __attribute__((aligned(32))) _Float16 pstage[4][16 * 32];
  const int wave = threadIdx.x >> 5;
  const int lane = threadIdx.x & 31;
  _Float16* ps = pstage[wave];

  const int t   = blockIdx.x * 4 + wave;
  const int qt  = t & 127;            // S/16 = 128 tiles
  const int h   = (t >> 7) & (H_ - 1);
  const int b   = t >> 11;
  const int kvh = h >> 2;             // N_REP = 4
  const int q0  = qt * 16;

  const int mrow = lane & 15;
  const int hi   = lane >> 4;
  const int koff = hi * 8;
  const int dkb  = hi * 8;
  const int nn   = lane & 15;

  // Q fragments (A-layout), two depth chunks covering HD=64.
  const _Float16* qrow = q16 + ((b * H_ + h) * S_ + q0 + mrow) * HD_;
  v16h qa0 = ldfrag16(qrow + koff,      qrow + 16 + koff);
  v16h qa1 = ldfrag16(qrow + 32 + koff, qrow + 48 + koff);

  const _Float16* kbase = k16 + (b * KH_ + kvh) * S_ * HD_;
  const _Float16* vbase = vT  + (b * KH_ + kvh) * HD_ * S_;

  v8f o0 = {}, o1 = {}, o2 = {}, o3 = {};
  float m_i[8], l_i[8];
  #pragma unroll
  for (int r = 0; r < 8; ++r) { m_i[r] = -3.0e30f; l_i[r] = 0.0f; }

  int klo = q0 - (WIN_ - 1); if (klo < 0) klo = 0; klo &= ~31;

  for (int kblk = klo; kblk <= q0 + 15; kblk += 32) {
    // ---- scores S = Q*K^T for two 16-key tiles ----
    const _Float16* krow0 = kbase + (kblk + nn) * HD_;
    const _Float16* krow1 = kbase + (kblk + 16 + nn) * HD_;
    v8f s0 = {}, s1 = {};
    v16h kf;
    kf = ldfrag16(krow0 + dkb,      krow0 + dkb + 16);      s0 = wmma16(qa0, kf, s0);
    kf = ldfrag16(krow0 + 32 + dkb, krow0 + 32 + dkb + 16); s0 = wmma16(qa1, kf, s0);
    kf = ldfrag16(krow1 + dkb,      krow1 + dkb + 16);      s1 = wmma16(qa0, kf, s1);
    kf = ldfrag16(krow1 + 32 + dkb, krow1 + 32 + dkb + 16); s1 = wmma16(qa1, kf, s1);

    // ---- mask + online softmax (per-row stats in 16-lane halves) ----
    const int key0 = kblk + nn, key1 = kblk + 16 + nn;
    float p0[8], p1[8], alpha[8];
    #pragma unroll
    for (int r = 0; r < 8; ++r) {
      int row = q0 + r + 8 * hi;
      float v0 = ((key0 <= row) && (key0 > row - WIN_)) ? s0[r] : -3.0e30f;
      float v1 = ((key1 <= row) && (key1 > row - WIN_)) ? s1[r] : -3.0e30f;
      float tm = fmaxf(v0, v1);
      #pragma unroll
      for (int m = 1; m < 16; m <<= 1) tm = fmaxf(tm, __shfl_xor(tm, m, 32));
      float mnew = fmaxf(m_i[r], tm);
      alpha[r] = __builtin_exp2f(m_i[r] - mnew);
      m_i[r] = mnew;
      float e0 = __builtin_exp2f(v0 - mnew);
      float e1 = __builtin_exp2f(v1 - mnew);
      p0[r] = e0; p1[r] = e1;
      float rs = e0 + e1;
      #pragma unroll
      for (int m = 1; m < 16; m <<= 1) rs += __shfl_xor(rs, m, 32);
      l_i[r] = l_i[r] * alpha[r] + rs;
    }
    #pragma unroll
    for (int r = 0; r < 8; ++r) {
      o0[r] *= alpha[r]; o1[r] *= alpha[r]; o2[r] *= alpha[r]; o3[r] *= alpha[r];
    }

    // ---- P: C-layout regs -> LDS -> A-layout fragment ----
    #pragma unroll
    for (int r = 0; r < 8; ++r) {
      ps[(r + 8 * hi) * 32 + nn]      = (_Float16)p0[r];
      ps[(r + 8 * hi) * 32 + 16 + nn] = (_Float16)p1[r];
    }
    // Same-wave LDS ops are in-order (DScnt); compiler inserts waits.
    v16h pa = ldfrag16(&ps[mrow * 32 + koff], &ps[mrow * 32 + 16 + koff]);

    // ---- O += P * V ; V-fragments contiguous thanks to vT layout ----
    const _Float16* vcol = vbase + nn * S_ + kblk + dkb;
    v16h vf;
    vf = ldfrag16(vcol,            vcol + 16);            o0 = wmma16(pa, vf, o0);
    vf = ldfrag16(vcol + 16 * S_,  vcol + 16 * S_ + 16);  o1 = wmma16(pa, vf, o1);
    vf = ldfrag16(vcol + 32 * S_,  vcol + 32 * S_ + 16);  o2 = wmma16(pa, vf, o2);
    vf = ldfrag16(vcol + 48 * S_,  vcol + 48 * S_ + 16);  o3 = wmma16(pa, vf, o3);
  }

  // ---- normalize and write attn output as [B*S, H*HD] fp32 ----
  #pragma unroll
  for (int r = 0; r < 8; ++r) {
    float inv = 1.0f / l_i[r];
    int row = (b * S_ + q0 + r + 8 * hi) * (H_ * HD_) + h * HD_;
    ao[row +  0 + nn] = o0[r] * inv;
    ao[row + 16 + nn] = o1[r] * inv;
    ao[row + 32 + nn] = o2[r] * inv;
    ao[row + 48 + nn] = o3[r] * inv;
  }
}

// ---------------------------------------------------------------------------
extern "C" void kernel_launch(void* const* d_in, const int* in_sizes, int n_in,
                              void* d_out, int out_size, void* d_ws, size_t ws_size,
                              hipStream_t stream) {
  (void)in_sizes; (void)n_in; (void)out_size; (void)ws_size;
  const float* x  = (const float*)d_in[0];
  const float* fc = (const float*)d_in[1];
  const float* fs = (const float*)d_in[2];
  const float* wq = (const float*)d_in[3];
  const float* wk = (const float*)d_in[4];
  const float* wv = (const float*)d_in[5];
  const float* wo = (const float*)d_in[6];
  const float* qw = (const float*)d_in[7];
  const float* kw = (const float*)d_in[8];
  float* out = (float*)d_out;

  char* ws = (char*)d_ws;
  float*    xq  = (float*)(ws);                               // 16 MiB
  float*    xk  = (float*)(ws + (size_t)16 * 1024 * 1024);    //  4 MiB
  float*    xv  = (float*)(ws + (size_t)20 * 1024 * 1024);    //  4 MiB
  _Float16* q16 = (_Float16*)(ws + (size_t)24 * 1024 * 1024); //  8 MiB
  _Float16* k16 = (_Float16*)(ws + (size_t)32 * 1024 * 1024); //  2 MiB
  _Float16* vT  = (_Float16*)(ws + (size_t)34 * 1024 * 1024); //  2 MiB
  float*    ao  = (float*)(ws + (size_t)36 * 1024 * 1024);    // 16 MiB

  const int M = B_ * S_;  // 4096

  // QKV projections
  k_gemm<<<dim3(M / BM, (H_ * HD_) / BN), 256, 0, stream>>>(x, wq, xq, M, H_ * HD_, D_);
  k_gemm<<<dim3(M / BM, (KH_ * HD_) / BN), 256, 0, stream>>>(x, wk, xk, M, KH_ * HD_, D_);
  k_gemm<<<dim3(M / BM, (KH_ * HD_) / BN), 256, 0, stream>>>(x, wv, xv, M, KH_ * HD_, D_);
  // RMSNorm + RoPE + transpose to head-major f16
  k_normrope<<<(B_ * S_ * H_ + B_ * S_ * KH_) / 8, 256, 0, stream>>>(
      xq, xk, fc, fs, qw, kw, q16, k16);
  // V convert + transpose to [B,KH,HD,S]
  k_vtrans<<<(B_ * KH_ * HD_ * S_) / 256, 256, 0, stream>>>(xv, vT);
  // Sliding-window flash attention
  k_attn<<<(B_ * H_ * (S_ / 16)) / 4, 128, 0, stream>>>(q16, k16, vT, ao);
  // Output projection
  k_gemm<<<dim3(M / BM, D_ / BN), 256, 0, stream>>>(ao, wo, out, M, D_, D_);
}
